// LSTMInitialisationPolicy_20083267076135
// MI455X (gfx1250) — matrix operand order repaired and checked by reading, not compiled, not run on hardware
//
#include <hip/hip_runtime.h>
#include <hip/hip_bf16.h>
#include <math.h>

// ---------------------------------------------------------------------------
// LSTM scan, 256 steps, H=4096.  Memory-bound: 3 gate matrices re-read every
// step.  Strategy: convert gates to bf16 (101 MB -> fits MI455X 192MB L2),
// persistent kernel, 1 grid barrier/step, WMMA bf16 mat-vec for the gates,
// z vector staged to LDS per step via CDNA5 async-to-LDS (ASYNCcnt).
// ---------------------------------------------------------------------------

#define H        4096
#define HP1      4097
#define KPAD     4128          // HP1 rounded up to multiple of 32
#define NCHUNK   (KPAD / 32)   // 129 k-chunks per dot product
#define ZCHUNK16 (KPAD / 8)    // 516 16-byte pieces of the z vector
#define TSTEPS   256
#define CDIM     1024
#define NBLK     256           // persistent grid: 256 blocks x 128 threads
#define NTHR     128

typedef __attribute__((ext_vector_type(16))) __bf16         v16bf;
typedef __attribute__((ext_vector_type(8)))  float          v8f;
typedef __attribute__((ext_vector_type(8)))  unsigned short ushort8;

union V16U { ushort8 h[2]; v16bf v; };
union V8FU { v8f v; float f[8]; };

static __device__ __forceinline__ unsigned short f2bf(float f) {
    union { float f; unsigned u; } v; v.f = f;
    unsigned r = v.u + 0x7FFFu + ((v.u >> 16) & 1u);   // round-nearest-even
    return (unsigned short)(r >> 16);
}
static __device__ __forceinline__ float bf2f(unsigned short b) {
    union { unsigned u; float f; } v; v.u = ((unsigned)b) << 16; return v.f;
}
static __device__ __forceinline__ float sigmoidf_(float x) {
    return 1.0f / (1.0f + __expf(-x));
}

// --- fp32 -> padded bf16 weight conversion (gates stacked: f,i,c rows) -----
__global__ void convert_kernel(const float* __restrict__ Wf,
                               const float* __restrict__ Wi,
                               const float* __restrict__ Wc,
                               unsigned short* __restrict__ wbf) {
    int k = blockIdx.x * 256 + threadIdx.x;     // guard to KPAD
    int r = blockIdx.y;                          // 0..3*H-1
    if (k >= KPAD) return;
    const float* W = (r < H) ? Wf : (r < 2 * H) ? Wi : Wc;
    int n = r & (H - 1);
    float v = (k < HP1) ? W[(size_t)n * HP1 + k] : 0.0f;
    wbf[(size_t)r * KPAD + k] = f2bf(v);
}

// --- init: x0 = l1_w . context + l1_b ; zero z double-buffer ; reset barrier
__global__ void init_kernel(const float* __restrict__ context,
                            const float* __restrict__ l1_w,
                            const float* __restrict__ l1_b,
                            unsigned short* __restrict__ zbuf,
                            unsigned* __restrict__ bar) {
    __shared__ float red[256];
    float p = 0.0f;
    for (int k = threadIdx.x; k < CDIM; k += 256) p += l1_w[k] * context[k];
    red[threadIdx.x] = p;
    __syncthreads();
    for (int s = 128; s > 0; s >>= 1) {
        if (threadIdx.x < (unsigned)s) red[threadIdx.x] += red[threadIdx.x + s];
        __syncthreads();
    }
    for (int i = threadIdx.x; i < 2 * KPAD; i += 256) zbuf[i] = 0;
    __syncthreads();
    if (threadIdx.x == 0) {
        zbuf[H] = f2bf(red[0] + l1_b[0]);   // x0 into z-buffer 0
        bar[0] = 0;                          // arrive counter
        bar[1] = 0;                          // generation
    }
}

// --- grid-wide sense barrier (all threads fence before arrival) ------------
static __device__ void grid_sync(unsigned* bar) {
    __threadfence();
    __syncthreads();
    if (threadIdx.x == 0) {
        volatile unsigned* vgen = bar + 1;
        unsigned g = *vgen;
        if (atomicAdd(bar, 1u) == NBLK - 1u) {
            *(volatile unsigned*)bar = 0;
            __threadfence();
            atomicAdd(bar + 1, 1u);
        } else {
            while (*vgen == g) __builtin_amdgcn_s_sleep(1);
        }
    }
    __syncthreads();
}

// --- persistent scan kernel ------------------------------------------------
__global__ __launch_bounds__(NTHR) void lstm_scan_kernel(
        const unsigned short* __restrict__ wbf,   // [3*H, KPAD] bf16
        unsigned short* __restrict__ zbuf,        // [2, KPAD]   bf16 (h..x..0)
        unsigned* __restrict__ bar,
        const float* __restrict__ u,
        const float* __restrict__ Wf_b,
        const float* __restrict__ Wi_b,
        const float* __restrict__ Wc_b,
        const float* __restrict__ Wo_w,
        const float* __restrict__ Wo_b,
        float* __restrict__ out)                  // [256 solution, 1 logp]
{
    __shared__ __align__(16) unsigned short zlds[KPAD];  // staged z vector
    __shared__ float gpre[48];    // f/i/c pre-activations for 16 units
    __shared__ float opart[32];   // o-gate partial sums (wave 3)
    __shared__ float cs[16];      // persistent cell state for this block

    const int lane = threadIdx.x & 31;
    const int wave = threadIdx.x >> 5;
    const int blk  = blockIdx.x;

    if (threadIdx.x < 16) cs[threadIdx.x] = 0.0f;
    float logp = 0.0f;

    // Wave 0..2: weight-row base pointer for WMMA A-tile (16 rows per block).
    const int m       = lane & 15;
    const int halfsel = lane >> 4;          // K-half select per A layout
    const unsigned short* wrow =
        wbf + ((size_t)(wave * H + blk * 16 + m)) * KPAD;

    const unsigned zlds_base = (unsigned)(size_t)(&zlds[0]);  // LDS offset

    __syncthreads();

    for (int t = 0; t < TSTEPS; ++t) {
        const unsigned short* zcur = zbuf + (size_t)(t & 1) * KPAD;
        unsigned short*       znxt = zbuf + (size_t)((t + 1) & 1) * KPAD;

        // --- stage z (8.25 KB) into LDS via async copy (ASYNCcnt) ----------
        for (int c = threadIdx.x; c < ZCHUNK16; c += NTHR) {
            unsigned           laddr = zlds_base + (unsigned)(c * 16);
            unsigned long long gaddr = (unsigned long long)(zcur + c * 8);
            asm volatile("global_load_async_to_lds_b128 %0, %1, off"
                         :: "v"(laddr), "v"(gaddr) : "memory");
        }
        asm volatile("s_wait_asynccnt 0" ::: "memory");
        __syncthreads();

        if (wave < 3) {
            // D(16x16) += A(16 rows x 32k of W) * B(z chunk bcast to 16 cols)
            // A from L2-resident global weight stream, B from LDS.
            v8f acc = {};
            for (int j = 0; j < NCHUNK; ++j) {
                const int k0 = j * 32;
                V16U A, B;
                A.h[0] = *(const ushort8*)(wrow + k0 + halfsel * 8);
                A.h[1] = *(const ushort8*)(wrow + k0 + halfsel * 8 + 16);
                B.h[0] = *(const ushort8*)(zlds + k0 + halfsel * 8);
                B.h[1] = *(const ushort8*)(zlds + k0 + halfsel * 8 + 16);
                acc = __builtin_amdgcn_wmma_f32_16x16x32_bf16(
                        false, A.v, false, B.v, (short)0, acc, false, false);
            }
            // All 16 columns identical; lane0 holds M=0..7, lane16 M=8..15.
            V8FU d; d.v = acc;
            if (lane == 0) {
                #pragma unroll
                for (int r = 0; r < 8; ++r) gpre[wave * 16 + r] = d.f[r];
            } else if (lane == 16) {
                #pragma unroll
                for (int r = 0; r < 8; ++r) gpre[wave * 16 + 8 + r] = d.f[r];
            }
        } else {
            // Wave 3: o-gate dot from LDS (redundant per block -> 1 sync/step)
            float p = 0.0f;
            for (int k = lane; k < HP1; k += 32)
                p += bf2f(zlds[k]) * Wo_w[k];
            opart[lane] = p;
        }
        __syncthreads();

        if (threadIdx.x < 16) {
            const int tl = threadIdx.x;
            const int n  = blk * 16 + tl;
            float osum = Wo_b[0];
            #pragma unroll
            for (int k = 0; k < 32; ++k) osum += opart[k];   // fixed order
            const float o  = sigmoidf_(osum);
            const float fg = sigmoidf_(gpre[tl]      + Wf_b[n]);
            const float ig = sigmoidf_(gpre[16 + tl] + Wi_b[n]);
            const float cg = tanhf   (gpre[32 + tl] + Wc_b[n]);
            const float cn = fg * cs[tl] + ig * cg;
            cs[tl] = cn;
            znxt[n] = f2bf(o * tanhf(cn));
            if (blk == 0 && tl == 0) {
                const float s = (u[t] < o) ? 1.0f : 0.0f;
                out[t] = s;
                logp += s * __logf(o) + (1.0f - s) * __logf(1.0f - o);
                znxt[H] = f2bf(s);               // x_{t+1} = s
            }
        }
        grid_sync(bar);
    }

    if (blk == 0 && threadIdx.x == 0) out[TSTEPS] = logp;
}

// ---------------------------------------------------------------------------
extern "C" void kernel_launch(void* const* d_in, const int* in_sizes, int n_in,
                              void* d_out, int out_size, void* d_ws, size_t ws_size,
                              hipStream_t stream) {
    const float* context = (const float*)d_in[0];
    const float* u       = (const float*)d_in[1];
    const float* l1_w    = (const float*)d_in[2];
    const float* l1_b    = (const float*)d_in[3];
    const float* Wf_w    = (const float*)d_in[4];
    const float* Wf_b    = (const float*)d_in[5];
    const float* Wi_w    = (const float*)d_in[6];
    const float* Wi_b    = (const float*)d_in[7];
    const float* Wc_w    = (const float*)d_in[8];
    const float* Wc_b    = (const float*)d_in[9];
    const float* Wo_w    = (const float*)d_in[10];
    const float* Wo_b    = (const float*)d_in[11];
    float* out = (float*)d_out;

    // Workspace layout (16B-aligned slices):
    //   [0)      bf16 gate weights, 3*H rows x KPAD  (~101.4 MB, L2-resident)
    //   [woff)   bf16 z double-buffer, 2 x KPAD
    //   [boff)   barrier counters (2 x u32)
    unsigned char* ws = (unsigned char*)d_ws;
    const size_t wbytes = (size_t)3 * H * KPAD * sizeof(unsigned short);
    unsigned short* wbf  = (unsigned short*)ws;
    unsigned short* zbuf = (unsigned short*)(ws + wbytes);
    unsigned*       bar  = (unsigned*)(ws + wbytes + 2 * KPAD * sizeof(unsigned short));

    // 1) fp32 -> padded bf16 weights (become L2-resident working set)
    dim3 cgrid((KPAD + 255) / 256, 3 * H);
    convert_kernel<<<cgrid, 256, 0, stream>>>(Wf_w, Wi_w, Wc_w, wbf);

    // 2) x0 = l1_w . context + l1_b ; zero z buffers ; reset barrier
    init_kernel<<<1, 256, 0, stream>>>(context, l1_w, l1_b, zbuf, bar);

    // 3) persistent 256-step scan
    lstm_scan_kernel<<<NBLK, NTHR, 0, stream>>>(
        wbf, zbuf, bar, u, Wf_b, Wi_b, Wc_b, Wo_w, Wo_b, out);
}
